// FP8Experts_90701119357755
// MI455X (gfx1250) — 8-bit matrix operand order repaired and checked by reading, compiled, not run on hardware
//
#include <hip/hip_runtime.h>
#include <hip/hip_bf16.h>


typedef __attribute__((ext_vector_type(16))) int   v16i;
typedef __attribute__((ext_vector_type(8)))  float v8f;

#define E_    8
#define H_    2048
#define NI    1408
#define T_    512
#define NBH   (H_ / 128)
#define NBI   (NI / 128)

#if defined(__has_builtin)
#if __has_builtin(__builtin_amdgcn_global_load_async_to_lds_b128) && \
    __has_builtin(__builtin_amdgcn_s_wait_asynccnt)
#define USE_ASYNC 1
#endif
#endif

#if defined(USE_ASYNC)
#define ASYNC_WAIT(n) __builtin_amdgcn_s_wait_asynccnt(n)
typedef int v4i_raw __attribute__((vector_size(16)));
typedef __attribute__((address_space(1))) v4i_raw* gv4i_p;
typedef __attribute__((address_space(3))) v4i_raw* lv4i_p;
#else
#define ASYNC_WAIT(n)
#endif

__device__ __forceinline__ unsigned int f32_to_e4m3(float f) {
  f = fminf(fmaxf(f, -448.0f), 448.0f);
  unsigned int b = __float_as_uint(f);
  unsigned int s = (b >> 24) & 0x80u;
  int exp = (int)((b >> 23) & 0xffu) - 127;
  unsigned int man = b & 0x7fffffu;
  if ((b & 0x7fffffffu) == 0u || exp < -10) return s;
  if (exp >= -6) {
    unsigned int m = man;
    unsigned int lsb = (m >> 20) & 1u;
    m += 0x7ffffu + lsb;
    if (m >> 23) { m = 0; exp++; }
    return s | (unsigned int)((exp + 7) << 3) | (m >> 20);
  }
  unsigned int m = man | 0x800000u;
  int shift = -6 - exp;
  unsigned int keep = m >> (20 + shift);
  unsigned int rmsb = (m >> (19 + shift)) & 1u;
  unsigned int sticky = (m & ((1u << (19 + shift)) - 1u)) != 0u;
  keep += (rmsb & (sticky | (keep & 1u)));
  return s | keep;
}

__global__ void k_quant_w(const float* __restrict__ w,
                          unsigned char* __restrict__ q, long long n4) {
  long long i = (long long)blockIdx.x * blockDim.x + threadIdx.x;
  if (i >= n4) return;
  float4 v = ((const float4*)w)[i];
  unsigned int c = f32_to_e4m3(v.x) | (f32_to_e4m3(v.y) << 8) |
                   (f32_to_e4m3(v.z) << 16) | (f32_to_e4m3(v.w) << 24);
  ((unsigned int*)q)[i] = c;
}

__global__ void k_quant_act(const float* __restrict__ x,
                            unsigned char* __restrict__ xq,
                            float* __restrict__ xs, int rows, int cols) {
  int lane = threadIdx.x & 31;
  int gw = (int)((blockIdx.x * blockDim.x + threadIdx.x) >> 5);
  int nb = cols >> 7;
  if (gw >= rows * nb) return;
  int row = gw / nb, blk = gw - row * nb;
  const float4* p = (const float4*)(x + (size_t)row * cols + blk * 128);
  float4 v = p[lane];
  float am = fmaxf(fmaxf(fabsf(v.x), fabsf(v.y)), fmaxf(fabsf(v.z), fabsf(v.w)));
#pragma unroll
  for (int o = 16; o > 0; o >>= 1) am = fmaxf(am, __shfl_xor(am, o, 32));
  float sc = fmaxf(am, 1e-12f) * (1.0f / 448.0f);
  float inv = 1.0f / sc;
  unsigned int c = f32_to_e4m3(v.x * inv) | (f32_to_e4m3(v.y * inv) << 8) |
                   (f32_to_e4m3(v.z * inv) << 16) | (f32_to_e4m3(v.w * inv) << 24);
  ((unsigned int*)(xq + (size_t)row * cols + blk * 128))[lane] = c;
  if (lane == 0) xs[(size_t)row * nb + blk] = sc;
}

__device__ __forceinline__ void tile_stage(unsigned char* dst_lds,
                                           const unsigned char* src,
                                           size_t stride, int tid) {
#pragma unroll
  for (int i = 0; i < 4; ++i) {
    int idx = tid + i * 256;
    int r = idx >> 3, c = (idx & 7) << 4;
#if defined(USE_ASYNC)
    __builtin_amdgcn_global_load_async_to_lds_b128(
        (gv4i_p)(src + (size_t)r * stride + c),
        (lv4i_p)(dst_lds + idx * 16), 0, 0);
#else
    *(uint4*)(dst_lds + idx * 16) = *(const uint4*)(src + (size_t)r * stride + c);
#endif
  }
}

__device__ __forceinline__ v16i load_a_frag(const unsigned char* As, int row, int hi) {
  v16i a;
#pragma unroll
  for (int v = 0; v < 16; ++v) {
    int k = (((v >> 1) & 3) << 4) + ((v >> 3) << 6) + (hi << 3) + ((v & 1) << 2);
    a[v] = *(const int*)(As + row * 128 + k);
  }
  return a;
}
__device__ __forceinline__ v16i load_b_frag(const unsigned char* Bs, int col, int hi) {
  v16i b;
#pragma unroll
  for (int v = 0; v < 16; ++v) {
    int k = (((v >> 1) & 3) << 4) + ((v >> 3) << 6) + (hi << 3) + ((v & 1) << 2);
    b[v] = *(const int*)(Bs + col * 128 + k);
  }
  return b;
}

__global__ __launch_bounds__(256) void k_gemm1(
    const unsigned char* __restrict__ xq, const float* __restrict__ xs,
    const unsigned char* __restrict__ wq, const float* __restrict__ wsc,
    unsigned char* __restrict__ hq, float* __restrict__ hs) {
  __shared__ __align__(16) unsigned char As[2][128 * 128];
  __shared__ __align__(16) unsigned char Bg[2][128 * 128];
  __shared__ __align__(16) unsigned char Bu[2][128 * 128];
  __shared__ float aScal[128];

  const int tid = threadIdx.x;
  const int lane = tid & 31, wave = tid >> 5;
  const int bx = blockIdx.x;
  const int t0 = blockIdx.y * 128;
  const int e = blockIdx.z;
  const int i0 = bx * 128;
  const int m0 = wave * 16;
  const int Ml = lane & 15;
  const int hi = lane >> 4;

  const unsigned char* srcA = xq + (size_t)t0 * H_;
  const unsigned char* srcG = wq + ((size_t)e * 2 * NI + i0) * H_;
  const unsigned char* srcU = wq + ((size_t)e * 2 * NI + NI + i0) * H_;

  v8f accG[8] = {};
  v8f accU[8] = {};

  tile_stage(As[0], srcA, H_, tid);
  tile_stage(Bg[0], srcG, H_, tid);
  tile_stage(Bu[0], srcU, H_, tid);

  for (int kb = 0; kb < NBH; ++kb) {
    const int cur = kb & 1;
    if (kb + 1 < NBH) {
      const size_t kn = (size_t)(kb + 1) * 128;
      tile_stage(As[cur ^ 1], srcA + kn, H_, tid);
      tile_stage(Bg[cur ^ 1], srcG + kn, H_, tid);
      tile_stage(Bu[cur ^ 1], srcU + kn, H_, tid);
      ASYNC_WAIT(12);
    } else {
      ASYNC_WAIT(0);
    }
    if (tid < 128) aScal[tid] = xs[(size_t)(t0 + tid) * NBH + kb];
    __syncthreads();

    const float sWg = wsc[((size_t)e * 2 * NBI + bx) * NBH + kb];
    const float sWu = wsc[((size_t)e * 2 * NBI + NBI + bx) * NBH + kb];

    v16i a = load_a_frag(As[cur], m0 + Ml, hi);
    float sa[8];
#pragma unroll
    for (int j = 0; j < 8; ++j) sa[j] = aScal[m0 + (hi << 3) + j];

#pragma unroll
    for (int nt = 0; nt < 8; ++nt) {
      v16i bg = load_b_frag(Bg[cur], nt * 16 + Ml, hi);
      v16i bu = load_b_frag(Bu[cur], nt * 16 + Ml, hi);
      v8f z = {};
      v8f dg = __builtin_amdgcn_wmma_f32_16x16x128_fp8_fp8(a, bg, (short)0, z, false, false);
      v8f du = __builtin_amdgcn_wmma_f32_16x16x128_fp8_fp8(a, bu, (short)0, z, false, false);
#pragma unroll
      for (int j = 0; j < 8; ++j) {
        accG[nt][j] += dg[j] * (sa[j] * sWg);
        accU[nt][j] += du[j] * (sa[j] * sWu);
      }
    }
    __syncthreads();
  }

#pragma unroll
  for (int j = 0; j < 8; ++j) {
    float h[8];
    float am = 0.0f;
#pragma unroll
    for (int nt = 0; nt < 8; ++nt) {
      float g = accG[nt][j], u = accU[nt][j];
      float v = (g / (1.0f + __expf(-g))) * u;
      h[nt] = v;
      am = fmaxf(am, fabsf(v));
    }
#pragma unroll
    for (int o = 1; o < 16; o <<= 1) am = fmaxf(am, __shfl_xor(am, o, 32));
    float sc = fmaxf(am, 1e-12f) * (1.0f / 448.0f);
    float inv = 1.0f / sc;
    int t = t0 + m0 + (hi << 3) + j;
#pragma unroll
    for (int nt = 0; nt < 8; ++nt)
      hq[((size_t)e * T_ + t) * NI + i0 + nt * 16 + Ml] =
          (unsigned char)f32_to_e4m3(h[nt] * inv);
    if (Ml == 0) hs[((size_t)e * T_ + t) * NBI + bx] = sc;
  }
}

__global__ __launch_bounds__(256) void k_gemm2(
    const unsigned char* __restrict__ hq, const float* __restrict__ hs,
    const unsigned char* __restrict__ wq, const float* __restrict__ wsc,
    const int* __restrict__ tki, const float* __restrict__ tkw,
    float* __restrict__ out) {
  __shared__ __align__(16) unsigned char As[2][128 * 128];
  __shared__ __align__(16) unsigned char Bs[2][128 * 128];
  __shared__ float hScal[128];
  __shared__ float combW[128 * 8];

  const int tid = threadIdx.x;
  const int lane = tid & 31, wave = tid >> 5;
  const int n0 = blockIdx.x * 128;
  const int t0 = blockIdx.y * 128;
  const int m0 = wave * 16;
  const int Ml = lane & 15;
  const int hi = lane >> 4;

  if (tid < 128) {
    int t = t0 + tid;
    int e0 = tki[t * 2 + 0], e1 = tki[t * 2 + 1];
    float w0 = tkw[t * 2 + 0], w1 = tkw[t * 2 + 1];
#pragma unroll
    for (int e = 0; e < E_; ++e)
      combW[tid * 8 + e] = (e0 == e ? w0 : 0.0f) + (e1 == e ? w1 : 0.0f);
  }

  v8f accT[8] = {};
  v8f accE[8] = {};
  const v8f zv = {};

  tile_stage(As[0], hq + (size_t)t0 * NI, NI, tid);
  tile_stage(Bs[0], wq + (size_t)n0 * NI, NI, tid);

  const int NIT = E_ * NBI;
  for (int idx = 0; idx < NIT; ++idx) {
    const int e = idx / NBI, kb = idx - e * NBI;
    const int cur = idx & 1;
    if (idx + 1 < NIT) {
      const int en = (idx + 1) / NBI, kbn = (idx + 1) - en * NBI;
      tile_stage(As[cur ^ 1],
                 hq + ((size_t)en * T_ + t0) * NI + (size_t)kbn * 128, NI, tid);
      tile_stage(Bs[cur ^ 1],
                 wq + ((size_t)en * H_ + n0) * NI + (size_t)kbn * 128, NI, tid);
      ASYNC_WAIT(8);
    } else {
      ASYNC_WAIT(0);
    }
    if (tid < 128) hScal[tid] = hs[((size_t)e * T_ + t0 + tid) * NBI + kb];
    __syncthreads();

    const float sW = wsc[((size_t)e * NBH + blockIdx.x) * NBI + kb];
    v16i a = load_a_frag(As[cur], m0 + Ml, hi);
    float sa[8];
#pragma unroll
    for (int j = 0; j < 8; ++j) sa[j] = hScal[m0 + (hi << 3) + j];

#pragma unroll
    for (int nt = 0; nt < 8; ++nt) {
      v16i b = load_b_frag(Bs[cur], nt * 16 + Ml, hi);
      v8f z = {};
      v8f d = __builtin_amdgcn_wmma_f32_16x16x128_fp8_fp8(a, b, (short)0, z, false, false);
#pragma unroll
      for (int j = 0; j < 8; ++j) accE[nt][j] += d[j] * (sa[j] * sW);
    }

    if (kb == NBI - 1) {
#pragma unroll
      for (int nt = 0; nt < 8; ++nt) {
#pragma unroll
        for (int j = 0; j < 8; ++j) {
          float c = combW[(m0 + (hi << 3) + j) * 8 + e];
          accT[nt][j] += c * accE[nt][j];
        }
        accE[nt] = zv;
      }
    }
    __syncthreads();
  }

#pragma unroll
  for (int nt = 0; nt < 8; ++nt)
#pragma unroll
    for (int j = 0; j < 8; ++j)
      out[(size_t)(t0 + m0 + (hi << 3) + j) * H_ + n0 + nt * 16 + Ml] = accT[nt][j];
}

extern "C" void kernel_launch(void* const* d_in, const int* in_sizes, int n_in,
                              void* d_out, int out_size, void* d_ws, size_t ws_size,
                              hipStream_t stream) {
  const float* x   = (const float*)d_in[0];
  const int*   tki = (const int*)d_in[1];
  const float* tkw = (const float*)d_in[2];
  const float* wgu = (const float*)d_in[3];
  const float* sgu = (const float*)d_in[4];
  const float* wd  = (const float*)d_in[5];
  const float* sd  = (const float*)d_in[6];
  float* out = (float*)d_out;

  unsigned char* ws = (unsigned char*)d_ws;
  size_t off = 0;
  auto alloc = [&](size_t bytes) -> unsigned char* {
    unsigned char* p = ws + off;
    off += (bytes + 255) & ~(size_t)255;
    return p;
  };
  unsigned char* wq_gu = alloc((size_t)E_ * 2 * NI * H_);
  unsigned char* wq_d  = alloc((size_t)E_ * H_ * NI);
  unsigned char* xq    = alloc((size_t)T_ * H_);
  unsigned char* hq    = alloc((size_t)E_ * T_ * NI);
  float* xsc = (float*)alloc((size_t)T_ * NBH * 4);
  float* hsc = (float*)alloc((size_t)E_ * T_ * NBI * 4);

  long long n1 = (long long)E_ * 2 * NI * H_ / 4;
  k_quant_w<<<(unsigned)((n1 + 255) / 256), 256, 0, stream>>>(wgu, wq_gu, n1);
  long long n2 = (long long)E_ * H_ * NI / 4;
  k_quant_w<<<(unsigned)((n2 + 255) / 256), 256, 0, stream>>>(wd, wq_d, n2);

  int nblk = T_ * NBH;
  k_quant_act<<<(nblk * 32 + 255) / 256, 256, 0, stream>>>(x, xq, xsc, T_, H_);

  k_gemm1<<<dim3(NBI, T_ / 128, E_), 256, 0, stream>>>(xq, xsc, wq_gu, sgu, hq, hsc);
  k_gemm2<<<dim3(NBH, T_ / 128), 256, 0, stream>>>(hq, hsc, wq_d, sd, tki, tkw, out);
}
